// ClassifierHead_31885837205766
// MI455X (gfx1250) — matrix operand order repaired and checked by reading, not compiled, not run on hardware
//
#include <hip/hip_runtime.h>
#include <hip/hip_bf16.h>

// ---------------------------------------------------------------------------
// ClassifierHead on gfx1250 (MI455X): 3 fused WMMA GEMM stages + softmax.
//   GEMM1: [4096,12544] x [12544,1024] fp32 -> f16 WMMA, fp32 acc -> BN+ReLU -> f16 ws
//   GEMM2: [4096,1024]  x [1024,1024]  (f16 A via async-to-LDS)    -> BN+ReLU -> f16 ws
//   GEMM3: [4096,1024]  x [1024,81] / [1024,324] -> +bias -> fp32 d_out
//   softmax over 81 classes (one wave32 per row)
// Block tile 128x128x32, 8 wave32s (4M x 2N), wave tile 32x64 = 2x4 WMMA accs.
// ---------------------------------------------------------------------------

typedef __attribute__((ext_vector_type(16))) _Float16 v16h;
typedef __attribute__((ext_vector_type(8)))  _Float16 v8h;
typedef __attribute__((ext_vector_type(4)))  _Float16 v4h;
typedef __attribute__((ext_vector_type(8)))  float    v8f;
typedef __attribute__((ext_vector_type(4)))  float    v4f;

#define BN_EPS 1e-3f

union Frag16 { v16h v; v8h h[2]; };

// CDNA5 async copy: global -> LDS, tracked by ASYNCcnt (ISA 15.18.3 op 98).
__device__ __forceinline__ void async_copy_b128_to_lds(unsigned lds_off, const void* g)
{
    asm volatile("global_load_async_to_lds_b128 %0, %1, off"
                 :: "v"(lds_off), "v"(g) : "memory");
}
__device__ __forceinline__ void wait_async_zero()
{
    asm volatile("s_wait_asynccnt 0x0" ::: "memory");
}

template<bool A_IS_F16, bool FUSE_BN_RELU>
__global__ __launch_bounds__(256)
void wmma_gemm_kernel(const void*  __restrict__ Aptr,   // [M,K] fp32 or f16, row-major
                      const float* __restrict__ B,      // [K,Ncols] fp32, row-major
                      const float* __restrict__ bias,
                      const float* __restrict__ gamma,
                      const float* __restrict__ beta,
                      const float* __restrict__ mean,
                      const float* __restrict__ var,
                      _Float16*    __restrict__ outH,   // f16 out (FUSE path)
                      float*       __restrict__ outF,   // fp32 out (plain path)
                      int Ncols, int K, int ldc)
{
    constexpr int BM = 128, BNT = 128, BK = 32;
    constexpr int LDA = BK + 8;   // f16 pad to break LDS bank conflicts
    constexpr int LDB = BK + 8;
    // As double-buffered for the async-copy (f16 A) path; f32 path uses buffer 0.
    __shared__ __align__(16) _Float16 As[2 * BM * LDA];
    __shared__ __align__(16) _Float16 Bs[BNT * LDB];

    const int tid  = threadIdx.x;
    const int lane = tid & 31;
    const int wave = tid >> 5;
    const int wm   = (wave & 3) * 32;    // wave M offset (4 waves in M)
    const int wn   = (wave >> 2) * 64;   // wave N offset (2 waves in N)
    const long bm  = (long)blockIdx.y * BM;
    const long bn  = (long)blockIdx.x * BNT;

    // ---- B staging: always fp32 global -> regs -> cvt -> LDS (n-major, K contig)
    const int b_n = tid & 127;           // column in tile (coalesced in n)
    const int b_k = (tid >> 7) * 16;     // 16 consecutive K per thread
    float b_stage[16];
    auto stage_B = [&](int k0) {
        long n = bn + b_n;
#pragma unroll
        for (int i = 0; i < 16; ++i) {
            long k = k0 + b_k + i;
            b_stage[i] = (n < Ncols) ? B[k * (long)Ncols + n] : 0.0f;
        }
    };
    auto commit_B = [&]() {
        v8h h0, h1;
#pragma unroll
        for (int i = 0; i < 8; ++i) { h0[i] = (_Float16)b_stage[i]; h1[i] = (_Float16)b_stage[8 + i]; }
        *(v8h*)(&Bs[b_n * LDB + b_k])     = h0;
        *(v8h*)(&Bs[b_n * LDB + b_k + 8]) = h1;
    };

    // ---- A path 1 (GEMM1): fp32 -> regs -> cvt f16 -> LDS
    const int a_m = tid >> 3;            // 0..31, 4 rows per thread (stride 32)
    const int a_k = (tid & 7) * 4;
    v4f a_stage[4];
    auto stage_A32 = [&](int k0) {
        const float* A32 = (const float*)Aptr;
#pragma unroll
        for (int r = 0; r < 4; ++r) {
            long m = bm + a_m + r * 32;
            a_stage[r] = *(const v4f*)(A32 + m * (long)K + k0 + a_k);
        }
    };
    auto commit_A32 = [&]() {
#pragma unroll
        for (int r = 0; r < 4; ++r) {
            v4h h;
#pragma unroll
            for (int i = 0; i < 4; ++i) h[i] = (_Float16)a_stage[r][i];
            *(v4h*)(&As[(a_m + r * 32) * LDA + a_k]) = h;
        }
    };

    // ---- A path 2 (GEMM2/3): f16 global -> LDS via async copy, double-buffered
    const int q_m = tid >> 1;            // 0..127: one row per thread pair
    const int q_k = (tid & 1) * 16;      // 16 f16 = 2 x b128 per thread
    auto issue_A16 = [&](int k0, int buf) {
        const _Float16* g = (const _Float16*)Aptr + (bm + q_m) * (long)K + k0 + q_k;
        unsigned l = (unsigned)(size_t)(&As[buf * BM * LDA + q_m * LDA + q_k]);
        async_copy_b128_to_lds(l,      (const void*)g);
        async_copy_b128_to_lds(l + 16, (const void*)(g + 8));
    };

    v8f acc[2][4] = {};

    auto compute = [&](const _Float16* Abase) {
        // Fragment gathers per ISA wave32 layouts:
        //  A 16x32: half-wave K base 0/8; V0-3 = kb..kb+7, V4-7 = kb+16..kb+23
        //  B 32x16: same K order as A (half-wave K base 0/8; kb..kb+7, kb+16..kb+23)
        const int ka = (lane >> 4) * 8;
        const int kb = (lane >> 4) * 16;
        Frag16 af[2], bf[4];
#pragma unroll
        for (int t = 0; t < 2; ++t) {
            const _Float16* ra = Abase + (wm + t * 16 + (lane & 15)) * LDA;
            af[t].h[0] = *(const v8h*)(ra + ka);
            af[t].h[1] = *(const v8h*)(ra + ka + 16);
        }
#pragma unroll
        for (int t = 0; t < 4; ++t) {
            const _Float16* rb = &Bs[(wn + t * 16 + (lane & 15)) * LDB];
            bf[t].h[0] = *(const v8h*)(rb + ka);
            bf[t].h[1] = *(const v8h*)(rb + ka + 16);
        }
#pragma unroll
        for (int mt = 0; mt < 2; ++mt)
#pragma unroll
            for (int nt = 0; nt < 4; ++nt)
                acc[mt][nt] = __builtin_amdgcn_wmma_f32_16x16x32_f16(
                    false, af[mt].v, false, bf[nt].v, (short)0, acc[mt][nt], false, false);
    };

    if constexpr (A_IS_F16) {
        stage_B(0);
        issue_A16(0, 0);
        for (int k0 = 0; k0 < K; k0 += BK) {
            const int buf = (k0 >> 5) & 1;
            __syncthreads();                  // all waves done reading Bs
            commit_B();
            wait_async_zero();                // this wave's A asyncs for `buf` landed
            __syncthreads();                  // everyone's commits + asyncs visible
            if (k0 + BK < K) { stage_B(k0 + BK); issue_A16(k0 + BK, buf ^ 1); }
            compute(&As[buf * BM * LDA]);
        }
    } else {
        stage_A32(0); stage_B(0);
        for (int k0 = 0; k0 < K; k0 += BK) {
            __syncthreads();
            commit_A32(); commit_B();
            __syncthreads();
            if (k0 + BK < K) { stage_A32(k0 + BK); stage_B(k0 + BK); }
            compute(As);
        }
    }

    // ---- Epilogue. C/D layout: lane L, vgpr r -> M = r + 8*(L/16), N = L%16.
#pragma unroll
    for (int nt = 0; nt < 4; ++nt) {
        long n = bn + wn + nt * 16 + (lane & 15);
        bool nok = n < Ncols;
        float bv = 0.f, sc = 1.f, mu = 0.f, bt = 0.f;
        if (nok) {
            bv = bias[n];
            if (FUSE_BN_RELU) {
                sc = gamma[n] * rsqrtf(var[n] + BN_EPS);
                mu = mean[n];
                bt = beta[n];
            }
        }
#pragma unroll
        for (int mt = 0; mt < 2; ++mt) {
            long mbase = bm + wm + mt * 16 + (lane >> 4) * 8;
#pragma unroll
            for (int r = 0; r < 8; ++r) {
                float v = acc[mt][nt][r] + bv;
                long  m = mbase + r;
                if (FUSE_BN_RELU) {
                    v = (v - mu) * sc + bt;
                    v = v > 0.f ? v : 0.f;
                    outH[m * (long)ldc + n] = (_Float16)v;
                } else if (nok) {
                    outF[m * (long)ldc + n] = v;
                }
            }
        }
    }
}

// One wave32 per row of 81 logits: lanes cover cols {l, l+32, l+64}.
__global__ __launch_bounds__(256)
void softmax81_kernel(const float* __restrict__ logits,
                      float* __restrict__ probs, int nrows)
{
    int row  = blockIdx.x * 8 + (threadIdx.x >> 5);
    int lane = threadIdx.x & 31;
    if (row >= nrows) return;
    const float* in  = logits + (long)row * 81;
    float*       out = probs  + (long)row * 81;

    float a = in[lane];
    float b = (lane + 32 < 81) ? in[lane + 32] : -3.0e38f;
    float c = (lane + 64 < 81) ? in[lane + 64] : -3.0e38f;
    float mx = fmaxf(a, fmaxf(b, c));
#pragma unroll
    for (int o = 16; o > 0; o >>= 1) mx = fmaxf(mx, __shfl_xor(mx, o, 32));
    float ea = __expf(a - mx);
    float eb = (lane + 32 < 81) ? __expf(b - mx) : 0.f;
    float ec = (lane + 64 < 81) ? __expf(c - mx) : 0.f;
    float s = ea + eb + ec;
#pragma unroll
    for (int o = 16; o > 0; o >>= 1) s += __shfl_xor(s, o, 32);
    float inv = 1.f / s;
    out[lane] = ea * inv;
    if (lane + 32 < 81) out[lane + 32] = eb * inv;
    if (lane + 64 < 81) out[lane + 64] = ec * inv;
}

extern "C" void kernel_launch(void* const* d_in, const int* in_sizes, int n_in,
                              void* d_out, int out_size, void* d_ws, size_t ws_size,
                              hipStream_t stream)
{
    const float* x   = (const float*)d_in[0];
    const float* w1  = (const float*)d_in[1];
    const float* b1  = (const float*)d_in[2];
    const float* g1  = (const float*)d_in[3];
    const float* be1 = (const float*)d_in[4];
    const float* m1  = (const float*)d_in[5];
    const float* v1  = (const float*)d_in[6];
    const float* w2  = (const float*)d_in[7];
    const float* b2  = (const float*)d_in[8];
    const float* g2  = (const float*)d_in[9];
    const float* be2 = (const float*)d_in[10];
    const float* m2  = (const float*)d_in[11];
    const float* v2  = (const float*)d_in[12];
    const float* wc  = (const float*)d_in[13];
    const float* bc  = (const float*)d_in[14];
    const float* wo  = (const float*)d_in[15];
    const float* bo  = (const float*)d_in[16];

    const int K1 = 7 * 7 * 256;                 // 12544
    const int N  = in_sizes[0] / K1;            // 4096
    const int H  = 1024;
    const int NC = 81;

    _Float16* y1 = (_Float16*)d_ws;             // [N,H] f16
    _Float16* y2 = y1 + (size_t)N * H;          // [N,H] f16
    float* logits = (float*)d_out;              // [N,81]
    float* probs  = logits + (size_t)N * NC;    // [N,81]
    float* offs   = probs  + (size_t)N * NC;    // [N,324]

    dim3 blk(256);

    // GEMM1 + BN + ReLU -> y1 (f16)
    wmma_gemm_kernel<false, true><<<dim3(H / 128, N / 128), blk, 0, stream>>>(
        (const void*)x, w1, b1, g1, be1, m1, v1, y1, nullptr, H, K1, H);

    // GEMM2 + BN + ReLU -> y2 (f16); A tiles via async-to-LDS
    wmma_gemm_kernel<true, true><<<dim3(H / 128, N / 128), blk, 0, stream>>>(
        (const void*)y1, w2, b2, g2, be2, m2, v2, y2, nullptr, H, H, H);

    // class logits -> d_out[0:N*81]
    wmma_gemm_kernel<true, false><<<dim3(1, N / 128), blk, 0, stream>>>(
        (const void*)y2, wc, bc, nullptr, nullptr, nullptr, nullptr,
        nullptr, logits, NC, H, NC);

    // box offsets -> d_out[2*N*81 : ], ldc = 324
    wmma_gemm_kernel<true, false><<<dim3((4 * NC + 127) / 128, N / 128), blk, 0, stream>>>(
        (const void*)y2, wo, bo, nullptr, nullptr, nullptr, nullptr,
        nullptr, offs, 4 * NC, H, 4 * NC);

    // softmax(logits) -> d_out[N*81 : 2*N*81]
    softmax81_kernel<<<dim3((N + 7) / 8), blk, 0, stream>>>(logits, probs, N);
}
